// CharRNN_69209103008411
// MI455X (gfx1250) — matrix operand order repaired and checked by reading, not compiled, not run on hardware
//
#include <hip/hip_runtime.h>
#include <hip/hip_bf16.h>
#include <math.h>

// Problem constants (match reference)
#define H_ 1024
#define B_ 32
#define T_ 128
#define V_ 10000
#define VPAD_ 10240   // V padded to a multiple of 64 n-tiles (zero rows appended)

typedef __attribute__((ext_vector_type(16))) __bf16 v16bf;
typedef __attribute__((ext_vector_type(8)))  __bf16 v8bf;
typedef __attribute__((ext_vector_type(8)))  float  v8f;

// ---------------------------------------------------------------------------
// gfx1250 async global->LDS path (ASYNCcnt-tracked). The builtin's pointer
// params are int4* in addrspace(1)/(3) (per hipcc diagnostic), so cast to
// exactly that. Guarded so the file still compiles (cooperative-copy fallback)
// on toolchains without the builtin.
// ---------------------------------------------------------------------------
#if defined(__has_builtin)
#if __has_builtin(__builtin_amdgcn_global_load_async_to_lds_b128)
#define HAVE_ASYNC_LDS 1
#endif
#endif

#ifdef HAVE_ASYNC_LDS
typedef int v4i_ __attribute__((vector_size(16)));
typedef __attribute__((address_space(1))) v4i_* glob_v4i_p;
typedef __attribute__((address_space(3))) v4i_* lds_v4i_p;

__device__ __forceinline__ void async_copy16(const __bf16* g, __bf16* l) {
  __builtin_amdgcn_global_load_async_to_lds_b128((glob_v4i_p)g, (lds_v4i_p)l, 0, 0);
}
__device__ __forceinline__ void async_wait0() {
#if __has_builtin(__builtin_amdgcn_s_wait_asynccnt)
  __builtin_amdgcn_s_wait_asynccnt(0);
#else
  asm volatile("s_wait_asynccnt 0" ::: "memory");
#endif
}
#endif

// ---------------------------------------------------------------------------
// Fragment loaders.
//
// A-fragment (16x32 bf16, M x K): lane l holds row m = l&15; its 16 values are
// K = h*8 + [0..8) and K = 16 + h*8 + [0..8) with h = l>>4  -> two contiguous
// 16B loads from a row-major [M][K] bf16 matrix.
//
// B-fragment (32x16 bf16, K x N): lane l holds column n = l&15; its 16 values
// are K = h*8 + [0..8) and K = 16 + h*8 + [0..8) -> two contiguous 16B loads
// from a [N][K] (transposed) bf16 matrix.
// ---------------------------------------------------------------------------
__device__ __forceinline__ v16bf load_a_frag(const __bf16* __restrict__ A, int ldk,
                                             int mtile, int kbase, int lane) {
  const __bf16* p = A + (size_t)(mtile * 16 + (lane & 15)) * ldk + kbase + ((lane >> 4) << 3);
  v8bf lo = *(const v8bf*)(p);
  v8bf hi = *(const v8bf*)(p + 16);
  v16bf a;
#pragma unroll
  for (int i = 0; i < 8; ++i) { a[i] = lo[i]; a[i + 8] = hi[i]; }
  return a;
}

// A-fragment sourced from an LDS-resident 16 x H_ tile (emits ds_load_b128)
__device__ __forceinline__ v16bf load_a_frag_lds(const __bf16* As, int kbase, int lane) {
  const __bf16* p = As + (size_t)(lane & 15) * H_ + kbase + ((lane >> 4) << 3);
  v8bf lo = *(const v8bf*)(p);
  v8bf hi = *(const v8bf*)(p + 16);
  v16bf a;
#pragma unroll
  for (int i = 0; i < 8; ++i) { a[i] = lo[i]; a[i + 8] = hi[i]; }
  return a;
}

__device__ __forceinline__ v16bf load_b_frag(const __bf16* __restrict__ Wt, int ldk,
                                             int ntile, int kbase, int lane) {
  const __bf16* p = Wt + (size_t)(ntile * 16 + (lane & 15)) * ldk + kbase + ((lane >> 4) << 3);
  v8bf lo = *(const v8bf*)(p);
  v8bf hi = *(const v8bf*)(p + 16);
  v16bf b;
#pragma unroll
  for (int i = 0; i < 8; ++i) { b[i] = lo[i]; b[i + 8] = hi[i]; }
  return b;
}

__device__ __forceinline__ v8f wmma_bf16(v16bf a, v16bf b, v8f c) {
  // v_wmma_f32_16x16x32_bf16: D = A x B + C, f32 accumulate
  return __builtin_amdgcn_wmma_f32_16x16x32_bf16(false, a, false, b, (short)0, c,
                                                 false, false);
}

// ---------------------------------------------------------------------------
// Prep kernels
// ---------------------------------------------------------------------------
__global__ void k_cvt_bf16(const float* __restrict__ in, __bf16* __restrict__ out, int n) {
  int i = blockIdx.x * blockDim.x + threadIdx.x;
  int stride = gridDim.x * blockDim.x;
  for (; i < n; i += stride) out[i] = (__bf16)in[i];
}

__global__ void k_zero_bf16(__bf16* __restrict__ p, int n) {
  int i = blockIdx.x * blockDim.x + threadIdx.x;
  int stride = gridDim.x * blockDim.x;
  for (; i < n; i += stride) p[i] = (__bf16)0.0f;
}

// in: [K][N] f32 row-major -> out: [N][K] bf16 row-major (LDS-tiled transpose)
__global__ void k_transpose_bf16(const float* __restrict__ in, __bf16* __restrict__ out,
                                 int K, int N) {
  __shared__ float tile[32][33];
  const int nb = blockIdx.x * 32;
  const int kb = blockIdx.y * 32;
  const int tx = threadIdx.x;  // 0..31
  const int ty = threadIdx.y;  // 0..7
#pragma unroll
  for (int i = 0; i < 32; i += 8)
    tile[ty + i][tx] = in[(size_t)(kb + ty + i) * N + (nb + tx)];
  __syncthreads();
#pragma unroll
  for (int i = 0; i < 32; i += 8)
    out[(size_t)(nb + ty + i) * K + (kb + tx)] = (__bf16)tile[tx][ty + i];
}

// X[t][b][:] = Ebf[idx[b][t]][:]   (X stored [T][B][H] so the per-t activation
// slice is a contiguous row-major [32][1024] matrix for the gate GEMM)
__global__ void k_embed(const int* __restrict__ idx, const __bf16* __restrict__ Ebf,
                        __bf16* __restrict__ X) {
  const int bt = blockIdx.x;
  const int t = bt % T_;
  const int b = bt / T_;
  const int row = idx[b * T_ + t];
  const uint2* src = (const uint2*)(Ebf + (size_t)row * H_);
  uint2* dst = (uint2*)(X + ((size_t)t * B_ + b) * H_);
  dst[threadIdx.x] = src[threadIdx.x];  // blockDim.x == H_/4 == 256
}

// ---------------------------------------------------------------------------
// GRU gate GEMM: G = sigmoid([A1 ; A2] @ Wg + bg), K = 2048, N = 2048.
// Epilogue: cols [0,1024)   -> r : write r * h_old  (bf16, row-major rH)
//           cols [1024,2048)-> u : write u          (f32, row-major U)
// One wave per 16x16 tile; block = 4 waves sharing mtile + consecutive ntiles
// (their A-fragment loads are identical addresses -> L0/L1 hits).
// ---------------------------------------------------------------------------
__global__ __launch_bounds__(128) void k_gru_gate(
    const __bf16* __restrict__ A1, const __bf16* __restrict__ A2,
    const __bf16* __restrict__ Wt, const float* __restrict__ bias,
    const __bf16* __restrict__ Hold, __bf16* __restrict__ rH,
    float* __restrict__ U) {
  const int lane  = threadIdx.x & 31;
  const int ntile = blockIdx.x * 4 + (threadIdx.x >> 5);  // 0..127
  const int mtile = blockIdx.y;                           // 0..1
  const size_t brow = (size_t)(ntile * 16 + (lane & 15)) * (2 * H_);
  v8f acc = {};
#pragma unroll 4
  for (int ks = 0; ks < 32; ++ks) {  // K = [0,1024): A1 (x or h0)
    __builtin_prefetch((const void*)(Wt + brow + ks * 32 + 512), 0, 3);  // near-scope
    v16bf a = load_a_frag(A1, H_, mtile, ks * 32, lane);
    v16bf b = load_b_frag(Wt, 2 * H_, ntile, ks * 32, lane);
    acc = wmma_bf16(a, b, acc);
  }
#pragma unroll 4
  for (int ks = 0; ks < 32; ++ks) {  // K = [1024,2048): A2 (h state)
    __builtin_prefetch((const void*)(Wt + brow + (ks + 32) * 32 + 512), 0, 3);
    v16bf a = load_a_frag(A2, H_, mtile, ks * 32, lane);
    v16bf b = load_b_frag(Wt, 2 * H_, ntile, (ks + 32) * 32, lane);
    acc = wmma_bf16(a, b, acc);
  }
  const int n  = lane & 15;
  const int hh = lane >> 4;
  const int gn = ntile * 16 + n;
  const float bv = bias[gn];
#pragma unroll
  for (int r = 0; r < 8; ++r) {
    const int m = mtile * 16 + r + 8 * hh;  // batch row
    float g = 1.0f / (1.0f + __expf(-(acc[r] + bv)));
    if (gn < H_) {  // wave-uniform branch (ntile uniform per wave)
      float hold = (float)Hold[m * H_ + gn];
      rH[m * H_ + gn] = (__bf16)(g * hold);
    } else {
      U[m * H_ + (gn - H_)] = g;
    }
  }
}

// ---------------------------------------------------------------------------
// GRU candidate GEMM + state update: c = tanh([A1 ; A2] @ Wc + bc),
// h_new = u*h_old + (1-u)*c.  K = 2048, N = 1024.
// Layer 1 additionally scatters h_new into Y at row b*T + t.
// ---------------------------------------------------------------------------
__global__ __launch_bounds__(128) void k_gru_cand(
    const __bf16* __restrict__ A1, const __bf16* __restrict__ A2,
    const __bf16* __restrict__ Wt, const float* __restrict__ bias,
    const float* __restrict__ U, __bf16* __restrict__ Hst,
    __bf16* __restrict__ Y, int t) {
  const int lane  = threadIdx.x & 31;
  const int ntile = blockIdx.x * 4 + (threadIdx.x >> 5);  // 0..63
  const int mtile = blockIdx.y;                           // 0..1
  const size_t brow = (size_t)(ntile * 16 + (lane & 15)) * (2 * H_);
  v8f acc = {};
#pragma unroll 4
  for (int ks = 0; ks < 32; ++ks) {  // K = [0,1024): x input
    __builtin_prefetch((const void*)(Wt + brow + ks * 32 + 512), 0, 3);  // near-scope
    v16bf a = load_a_frag(A1, H_, mtile, ks * 32, lane);
    v16bf b = load_b_frag(Wt, 2 * H_, ntile, ks * 32, lane);
    acc = wmma_bf16(a, b, acc);
  }
#pragma unroll 4
  for (int ks = 0; ks < 32; ++ks) {  // K = [1024,2048): r*h
    __builtin_prefetch((const void*)(Wt + brow + (ks + 32) * 32 + 512), 0, 3);
    v16bf a = load_a_frag(A2, H_, mtile, ks * 32, lane);
    v16bf b = load_b_frag(Wt, 2 * H_, ntile, (ks + 32) * 32, lane);
    acc = wmma_bf16(a, b, acc);
  }
  const int n  = lane & 15;
  const int hh = lane >> 4;
  const int gn = ntile * 16 + n;
  const float bv = bias[gn];
#pragma unroll
  for (int r = 0; r < 8; ++r) {
    const int m = mtile * 16 + r + 8 * hh;  // batch row
    float c = tanhf(acc[r] + bv);
    float u = U[m * H_ + gn];
    float hold = (float)Hst[m * H_ + gn];
    float hn = u * hold + (1.0f - u) * c;
    Hst[m * H_ + gn] = (__bf16)hn;
    if (Y) Y[((size_t)m * T_ + t) * H_ + gn] = (__bf16)hn;  // row = b*T + t
  }
}

// ---------------------------------------------------------------------------
// Logits: out[4096,10000] = Y[4096,1024] @ E^T + softmax_b.
// E (bf16, padded to 10240 zero rows) IS the [N][K] transposed-B layout.
// Block = 256 threads = 8 waves, one mtile; the 32 KB A tile is staged in LDS
// (async global->LDS when available) and shared by 8 waves x 4 n-tiles each.
// ---------------------------------------------------------------------------
__device__ __forceinline__ void store_logits_tile(v8f acc, int mtile, int ntile,
                                                  const float* __restrict__ sb,
                                                  float* __restrict__ out, int lane) {
  const int n  = lane & 15;
  const int hh = lane >> 4;
  const int gn = ntile * 16 + n;
  if (gn < V_) {
    const float bv = sb[gn];
#pragma unroll
    for (int r = 0; r < 8; ++r)
      out[(size_t)(mtile * 16 + r + 8 * hh) * V_ + gn] = acc[r] + bv;
  }
}

__global__ __launch_bounds__(256) void k_logits(
    const __bf16* __restrict__ Y, const __bf16* __restrict__ Ebf,
    const float* __restrict__ sb, float* __restrict__ out) {
  __shared__ __bf16 As[16 * H_];  // 32 KB: one 16-row A tile
  const int lane  = threadIdx.x & 31;
  const int wave  = threadIdx.x >> 5;
  const int mtile = blockIdx.y;                         // 0..255
  const int nt0   = (blockIdx.x * 8 + wave) * 4;        // 4 n-tiles per wave (640 total)

  const __bf16* Ag = Y + (size_t)mtile * 16 * H_;
#ifdef HAVE_ASYNC_LDS
#pragma unroll
  for (int i = 0; i < 8; ++i) {                 // 2048 x 16B chunks / 256 threads
    const int c = i * 256 + threadIdx.x;
    async_copy16(Ag + c * 8, As + c * 8);
  }
  async_wait0();
  __syncthreads();
#else
  {
    const uint4* s = (const uint4*)Ag;
    uint4* d = (uint4*)As;
    for (int i = threadIdx.x; i < 16 * H_ / 8; i += 256) d[i] = s[i];
    __syncthreads();
  }
#endif

  v8f acc0 = {}, acc1 = {}, acc2 = {}, acc3 = {};
#pragma unroll 2
  for (int ks = 0; ks < H_ / 32; ++ks) {
    v16bf a  = load_a_frag_lds(As, ks * 32, lane);
    v16bf b0 = load_b_frag(Ebf, H_, nt0 + 0, ks * 32, lane);
    v16bf b1 = load_b_frag(Ebf, H_, nt0 + 1, ks * 32, lane);
    v16bf b2 = load_b_frag(Ebf, H_, nt0 + 2, ks * 32, lane);
    v16bf b3 = load_b_frag(Ebf, H_, nt0 + 3, ks * 32, lane);
    acc0 = wmma_bf16(a, b0, acc0);
    acc1 = wmma_bf16(a, b1, acc1);
    acc2 = wmma_bf16(a, b2, acc2);
    acc3 = wmma_bf16(a, b3, acc3);
  }
  store_logits_tile(acc0, mtile, nt0 + 0, sb, out, lane);
  store_logits_tile(acc1, mtile, nt0 + 1, sb, out, lane);
  store_logits_tile(acc2, mtile, nt0 + 2, sb, out, lane);
  store_logits_tile(acc3, mtile, nt0 + 3, sb, out, lane);
}

// ---------------------------------------------------------------------------
// Host launcher
// ---------------------------------------------------------------------------
static inline size_t align256(size_t x) { return (x + 255) & ~(size_t)255; }

extern "C" void kernel_launch(void* const* d_in, const int* in_sizes, int n_in,
                              void* d_out, int out_size, void* d_ws, size_t ws_size,
                              hipStream_t stream) {
  const int*   idx = (const int*)  d_in[0];
  const float* E   = (const float*)d_in[1];
  const float* Wg0 = (const float*)d_in[2];
  const float* bg0 = (const float*)d_in[3];
  const float* Wc0 = (const float*)d_in[4];
  const float* bc0 = (const float*)d_in[5];
  const float* Wg1 = (const float*)d_in[6];
  const float* bg1 = (const float*)d_in[7];
  const float* Wc1 = (const float*)d_in[8];
  const float* bc1 = (const float*)d_in[9];
  const float* sb  = (const float*)d_in[10];
  float* out = (float*)d_out;
  (void)in_sizes; (void)n_in; (void)out_size; (void)ws_size;

  // Workspace carve-out (~64 MB total)
  char* w = (char*)d_ws;
  auto take = [&](size_t bytes) { char* p = w; w += align256(bytes); return p; };
  __bf16* Ebf  = (__bf16*)take((size_t)VPAD_ * H_ * 2);     // embedding bf16, zero-padded rows
  __bf16* X    = (__bf16*)take((size_t)T_ * B_ * H_ * 2);   // embedded inputs [T][B][H]
  __bf16* Wg0t = (__bf16*)take((size_t)2 * H_ * 2 * H_ * 2);// [2H][2H] transposed
  __bf16* Wc0t = (__bf16*)take((size_t)H_ * 2 * H_ * 2);    // [H][2H] transposed
  __bf16* Wg1t = (__bf16*)take((size_t)2 * H_ * 2 * H_ * 2);
  __bf16* Wc1t = (__bf16*)take((size_t)H_ * 2 * H_ * 2);
  __bf16* Hs0  = (__bf16*)take((size_t)B_ * H_ * 2);        // layer-0 state
  __bf16* Hs1  = (__bf16*)take((size_t)B_ * H_ * 2);        // layer-1 state
  __bf16* rH   = (__bf16*)take((size_t)B_ * H_ * 2);        // r * h scratch
  float*  U    = (float*) take((size_t)B_ * H_ * 4);        // update gate scratch
  __bf16* Yb   = (__bf16*)take((size_t)B_ * T_ * H_ * 2);   // layer-1 outputs [B*T][H]

  // 1) bf16 conversions / weight transposes (weights then live in L2 as bf16)
  k_cvt_bf16<<<2048, 256, 0, stream>>>(E, Ebf, V_ * H_);
  k_zero_bf16<<<240, 256, 0, stream>>>(Ebf + (size_t)V_ * H_, (VPAD_ - V_) * H_);
  dim3 tb(32, 8);
  k_transpose_bf16<<<dim3(2 * H_ / 32, 2 * H_ / 32), tb, 0, stream>>>(Wg0, Wg0t, 2 * H_, 2 * H_);
  k_transpose_bf16<<<dim3(H_ / 32,     2 * H_ / 32), tb, 0, stream>>>(Wc0, Wc0t, 2 * H_, H_);
  k_transpose_bf16<<<dim3(2 * H_ / 32, 2 * H_ / 32), tb, 0, stream>>>(Wg1, Wg1t, 2 * H_, 2 * H_);
  k_transpose_bf16<<<dim3(H_ / 32,     2 * H_ / 32), tb, 0, stream>>>(Wc1, Wc1t, 2 * H_, H_);

  // 2) embedding gather + zero states
  k_embed<<<B_ * T_, H_ / 4, 0, stream>>>(idx, Ebf, X);
  k_zero_bf16<<<8, 256, 0, stream>>>(Hs0, B_ * H_);
  k_zero_bf16<<<8, 256, 0, stream>>>(Hs1, B_ * H_);

  // 3) recurrence (sequential over T; 4 WMMA kernels per timestep)
  for (int t = 0; t < T_; ++t) {
    const __bf16* Xt = X + (size_t)t * B_ * H_;
    // layer 0: gates from [x_t ; h0], candidate from [x_t ; r*h0]
    k_gru_gate<<<dim3(32, 2), 128, 0, stream>>>(Xt, Hs0, Wg0t, bg0, Hs0, rH, U);
    k_gru_cand<<<dim3(16, 2), 128, 0, stream>>>(Xt, rH, Wc0t, bc0, U, Hs0, (__bf16*)nullptr, t);
    // layer 1: input is the freshly-updated h0
    k_gru_gate<<<dim3(32, 2), 128, 0, stream>>>(Hs0, Hs1, Wg1t, bg1, Hs1, rH, U);
    k_gru_cand<<<dim3(16, 2), 128, 0, stream>>>(Hs0, rH, Wc1t, bc1, U, Hs1, Yb, t);
  }

  // 4) tied-softmax logits GEMM (+bias), f32 output
  k_logits<<<dim3(VPAD_ / 16 / 32, (B_ * T_) / 16), 256, 0, stream>>>(Yb, Ebf, sb, out);
}
